// MultiHeadSelfAttention_30975304139526
// MI455X (gfx1250) — compile-verified, matrix operand order checked by reading, not run on hardware
//
#include <hip/hip_runtime.h>

// ---------- vector types ----------
typedef __attribute__((ext_vector_type(16))) _Float16 v16h;
typedef __attribute__((ext_vector_type(8)))  _Float16 v8h;
typedef __attribute__((ext_vector_type(8)))  float    v8f;
typedef __attribute__((ext_vector_type(4)))  unsigned int u32x4;
typedef __attribute__((ext_vector_type(8)))  int      i32x8;
typedef __attribute__((ext_vector_type(4)))  int      i32x4;

#if __has_builtin(__builtin_amdgcn_tensor_load_to_lds) && __has_builtin(__builtin_amdgcn_s_wait_tensorcnt)
#define USE_TDM 1
#else
#define USE_TDM 0
#endif

union V16U { v16h v; v8h h[2]; };

__device__ __forceinline__ v16h ldfrag(const _Float16* p0, const _Float16* p1) {
  // ISA 7.12.2 16-bit A/B fragment: elems 0..7 from p0 (16B), elems 8..15 from p1 (16B)
  V16U u;
  u.h[0] = *(const v8h*)p0;
  u.h[1] = *(const v8h*)p1;
  return u.v;
}

__device__ __forceinline__ v8f wmma_f16(v16h a, v16h b, v8f c) {
  return __builtin_amdgcn_wmma_f32_16x16x32_f16(false, a, false, b, (short)0, c, false, false);
}

// xor-butterfly max over 16-lane halves using DS swizzle immediates
__device__ __forceinline__ float xor_max16(float v) {
  v = fmaxf(v, __int_as_float(__builtin_amdgcn_ds_swizzle(__float_as_int(v), 0x041f))); // xor 1
  v = fmaxf(v, __int_as_float(__builtin_amdgcn_ds_swizzle(__float_as_int(v), 0x081f))); // xor 2
  v = fmaxf(v, __int_as_float(__builtin_amdgcn_ds_swizzle(__float_as_int(v), 0x101f))); // xor 4
  v = fmaxf(v, __int_as_float(__builtin_amdgcn_ds_swizzle(__float_as_int(v), 0x201f))); // xor 8
  return v;
}

#if USE_TDM
// TDM 2D tile load: 64x64 f16 tile, row stride `stride_elems` in memory,
// LDS rows padded 128B data + 16B pad (matches [64][72] _Float16 layout).
// D# packing per ISA 8.3/8.4 (group0: count/lds/global/type; group1: dims).
__device__ __forceinline__ void tdm_load_2d(unsigned lds_off, const _Float16* gptr,
                                            unsigned stride_elems) {
  unsigned long long ga = (unsigned long long)(uintptr_t)gptr;
  u32x4 g0;
  g0[0] = 1u;                                             // count=1 (valid), user mode
  g0[1] = lds_off;                                        // lds_addr (bytes)
  g0[2] = (unsigned)(ga & 0xffffffffu);                   // global_addr[31:0]
  g0[3] = (unsigned)((ga >> 32) & 0x1ffffffu) | (2u << 30); // global_addr[56:32] | type=2
  i32x8 g1;
  g1[0] = (int)((1u << 16)      // data_size = 1 (2 bytes)
              | (1u << 20)      // pad_enable
              | (4u << 22)      // pad_interval: 8<<4 = 128B between pads
              | (3u << 25));    // pad_amount: 4 DWORDs = 16B
  g1[1] = (int)(64u << 16);     // tensor_dim0 = 64 (low 16 in [31:16])
  g1[2] = (int)(64u << 16);     // tensor_dim1 = 64 (low 16 in [31:16])
  g1[3] = (int)(64u << 16);     // tile_dim0 = 64 ([31:16])
  g1[4] = 64;                   // tile_dim1 = 64 ([15:0]); tile_dim2 = 0
  g1[5] = (int)stride_elems;    // tensor_dim0_stride low 32
  g1[6] = 0;                    // stride hi / dim1_stride low
  g1[7] = 0;
  i32x4 g2 = {0, 0, 0, 0}, g3 = {0, 0, 0, 0};
#if __clang_major__ >= 23
  i32x8 g4 = {0, 0, 0, 0, 0, 0, 0, 0};
  __builtin_amdgcn_tensor_load_to_lds(g0, g1, g2, g3, g4, 0);
#else
  __builtin_amdgcn_tensor_load_to_lds(g0, g1, g2, g3, 0);
#endif
}
#endif

// ---------- problem constants ----------
#define EMBD   1024
#define HEADS  16
#define HDIM   64
#define SEQ    2048
#define BATCH  4
#define ROWS   (BATCH * SEQ)        // 8192
#define BH     (BATCH * HEADS)      // 64
#define INV_SCALE 0.125f            // 1/sqrt(64)

// =====================================================================
// fp32 -> f16 convert (grid-stride)
// =====================================================================
__global__ void convert_f32_f16(const float* __restrict__ src,
                                _Float16* __restrict__ dst, int n) {
  int i = blockIdx.x * blockDim.x + threadIdx.x;
  int stride = gridDim.x * blockDim.x;
  for (; i < n; i += stride) dst[i] = (_Float16)src[i];
}

// =====================================================================
// pack Wq/Wk/Wv -> stacked f16 [3072][1024], Wo -> f16, biases -> f32[3072]
// =====================================================================
__global__ void pack_weights(const float* __restrict__ Wq, const float* __restrict__ Wk,
                             const float* __restrict__ Wv, const float* __restrict__ Wo,
                             const float* __restrict__ bq, const float* __restrict__ bk,
                             const float* __restrict__ bv,
                             _Float16* __restrict__ Wqkv, _Float16* __restrict__ Woh,
                             float* __restrict__ bqkv) {
  int i = blockIdx.x * blockDim.x + threadIdx.x;
  int stride = gridDim.x * blockDim.x;
  const int NW = EMBD * EMBD;
  for (int k = i; k < NW; k += stride) {
    Wqkv[k]          = (_Float16)Wq[k];
    Wqkv[NW + k]     = (_Float16)Wk[k];
    Wqkv[2 * NW + k] = (_Float16)Wv[k];
    Woh[k]           = (_Float16)Wo[k];
  }
  for (int k = i; k < EMBD; k += stride) {
    bqkv[k]            = bq[k];
    bqkv[EMBD + k]     = bk[k];
    bqkv[2 * EMBD + k] = bv[k];
  }
}

// =====================================================================
// Tiled WMMA GEMM: C[M x N] = A[M x 1024] * B[N x 1024]^T  (K contig both)
// Block tile 128x128, 8 waves (4M x 2N), wave tile 32x64 (2x4 WMMA tiles),
// BK=32, double-buffered LDS. MODE 0: scatter to Q/K/Vt with bias (+Q scale).
// MODE 1: add bias, write fp32 to Cout.
// =====================================================================
template <int MODE>
__global__ __launch_bounds__(256) void gemm_wmma(
    const _Float16* __restrict__ A, const _Float16* __restrict__ B,
    const float* __restrict__ bias,
    _Float16* __restrict__ Qh, _Float16* __restrict__ Kh, _Float16* __restrict__ Vt,
    float* __restrict__ Cout) {
  __shared__ __align__(16) _Float16 As[2][128][40];  // +8 half pad -> conflict-free b128
  __shared__ __align__(16) _Float16 Bs[2][128][40];

  const int tid  = threadIdx.x;
  const int lane = tid & 31;
  const int wid  = tid >> 5;
  const int wm   = wid & 3;
  const int wn   = wid >> 2;
  const int blockM = blockIdx.x * 128;
  const int blockN = blockIdx.y * 128;

  const int c0 = tid, c1 = tid + 256;
  const int ar0 = c0 >> 2, acol0 = (c0 & 3) * 8;
  const int ar1 = c1 >> 2, acol1 = (c1 & 3) * 8;

  const _Float16* Abase = A + (long)blockM * EMBD;
  const _Float16* Bbase = B + (long)blockN * EMBD;

  v8h ra0, ra1, rb0, rb1;
  auto load_tile = [&](int kt) {
    const int ko = kt * 32;
    ra0 = *(const v8h*)(Abase + (long)ar0 * EMBD + ko + acol0);
    ra1 = *(const v8h*)(Abase + (long)ar1 * EMBD + ko + acol1);
    rb0 = *(const v8h*)(Bbase + (long)ar0 * EMBD + ko + acol0);
    rb1 = *(const v8h*)(Bbase + (long)ar1 * EMBD + ko + acol1);
  };
  auto store_tile = [&](int buf) {
    *(v8h*)&As[buf][ar0][acol0] = ra0;
    *(v8h*)&As[buf][ar1][acol1] = ra1;
    *(v8h*)&Bs[buf][ar0][acol0] = rb0;
    *(v8h*)&Bs[buf][ar1][acol1] = rb1;
  };

  v8f acc[2][4];
#pragma unroll
  for (int tm = 0; tm < 2; ++tm)
#pragma unroll
    for (int tn = 0; tn < 4; ++tn) acc[tm][tn] = {};

  load_tile(0);
  store_tile(0);

  const int r  = lane & 15;
  const int kg = lane >> 4;
  const int NK = EMBD / 32;

  for (int kt = 0; kt < NK; ++kt) {
    const int cur = kt & 1;
    if (kt + 1 < NK) {
      load_tile(kt + 1);
      if (kt + 2 < NK)
        __builtin_prefetch(Abase + (long)ar0 * EMBD + (kt + 2) * 32 + acol0, 0, 1);
    }
    __syncthreads();

    v16h a0 = ldfrag(&As[cur][wm * 32 + r][kg * 8],      &As[cur][wm * 32 + r][16 + kg * 8]);
    v16h a1 = ldfrag(&As[cur][wm * 32 + 16 + r][kg * 8], &As[cur][wm * 32 + 16 + r][16 + kg * 8]);
    v16h b[4];
#pragma unroll
    for (int tn = 0; tn < 4; ++tn)
      b[tn] = ldfrag(&Bs[cur][wn * 64 + tn * 16 + r][kg * 8],
                     &Bs[cur][wn * 64 + tn * 16 + r][16 + kg * 8]);

#pragma unroll
    for (int tn = 0; tn < 4; ++tn) {
      acc[0][tn] = wmma_f16(a0, b[tn], acc[0][tn]);
      acc[1][tn] = wmma_f16(a1, b[tn], acc[1][tn]);
    }

    if (kt + 1 < NK) {
      __syncthreads();
      store_tile((kt + 1) & 1);
    }
  }

  // ---------------- epilogue ----------------
  const int hh = lane >> 4;
#pragma unroll
  for (int tm = 0; tm < 2; ++tm) {
#pragma unroll
    for (int tn = 0; tn < 4; ++tn) {
      v8f a = acc[tm][tn];
      const int gN = blockN + wn * 64 + tn * 16 + r;
      const float bv = bias[gN];
#pragma unroll
      for (int i = 0; i < 8; ++i) {
        const int gM = blockM + wm * 32 + tm * 16 + i + 8 * hh;
        float v = a[i] + bv;
        if (MODE == 0) {
          const int which = gN >> 10;        // 0=Q 1=K 2=V
          const int e = gN & 1023;
          const int h = e >> 6;
          const int d = e & 63;
          const int b2 = gM >> 11;
          const int s = gM & 2047;
          const long bh = (long)b2 * HEADS + h;
          if (which == 0)      Qh[(bh * SEQ + s) * HDIM + d] = (_Float16)(v * INV_SCALE);
          else if (which == 1) Kh[(bh * SEQ + s) * HDIM + d] = (_Float16)v;
          else                 Vt[(bh * HDIM + d) * SEQ + s] = (_Float16)v;
        } else {
          Cout[(long)gM * EMBD + gN] = v;
        }
      }
    }
  }
}

// =====================================================================
// Flash-attention: grid = BH * (SEQ/128); block = 256 (8 waves).
// Key blocks of 64; K/V tiles staged into LDS via the Tensor Data Mover
// (double buffered, TENSORcnt-tracked), falling back to reg staging if the
// TDM builtin is unavailable. Row-sum via WMMA; row-max via ds_swizzle.
// =====================================================================
__global__ __launch_bounds__(256) void attn_flash(
    const _Float16* __restrict__ Qh, const _Float16* __restrict__ Kh,
    const _Float16* __restrict__ Vt, _Float16* __restrict__ Ch) {
  __shared__ __align__(16) _Float16 Ks[2][64][72];   // keys x d      (18 KB)
  __shared__ __align__(16) _Float16 Vs[2][64][72];   // d x keys      (18 KB)
  __shared__ __align__(16) _Float16 Pl[8][16][72];   // per-wave P    (18 KB)

  const int tid  = threadIdx.x;
  const int lane = tid & 31;
  const int w    = tid >> 5;
  const int qt   = blockIdx.x & 15;
  const int bh   = blockIdx.x >> 4;
  const int b    = bh >> 4;
  const int head = bh & 15;

  const int r  = lane & 15;
  const int kg = lane >> 4;
  const int hh = kg;
  const int qbase = qt * 128 + w * 16;

  const _Float16* Kbh = Kh + (long)bh * SEQ * HDIM;
  const _Float16* Vbh = Vt + (long)bh * HDIM * SEQ;

  // Q fragments (Q pre-scaled by 1/sqrt(d))
  const _Float16* Qp = Qh + ((long)bh * SEQ + qbase + r) * HDIM;
  const v16h qf0 = ldfrag(Qp + kg * 8,      Qp + 16 + kg * 8);
  const v16h qf1 = ldfrag(Qp + 32 + kg * 8, Qp + 48 + kg * 8);

  // ones B-fragment for WMMA row-sum
  v16h ones;
#pragma unroll
  for (int e = 0; e < 16; ++e) ones[e] = (_Float16)1.0f;

  const int NB = SEQ / 64;  // 32 key blocks

#if USE_TDM
  const unsigned ks_off0 = (unsigned)(uintptr_t)&Ks[0][0][0];
  const unsigned ks_off1 = (unsigned)(uintptr_t)&Ks[1][0][0];
  const unsigned vs_off0 = (unsigned)(uintptr_t)&Vs[0][0][0];
  const unsigned vs_off1 = (unsigned)(uintptr_t)&Vs[1][0][0];
  auto tdm_pair = [&](int buf, int kb) {
    const int kbase = kb * 64;
    tdm_load_2d(buf ? ks_off1 : ks_off0, Kbh + (long)kbase * HDIM, HDIM); // K: 64 keys x 64 d
    tdm_load_2d(buf ? vs_off1 : vs_off0, Vbh + kbase, SEQ);               // V: 64 d x 64 keys
  };
  if (w == 0) tdm_pair(0, 0);
#else
  // fallback: global->VGPR->LDS staging (2+2 16B chunks per thread)
  const int c0 = tid, c1 = tid + 256;
  const int sr0 = c0 >> 3, sc0 = (c0 & 7) * 8;
  const int sr1 = c1 >> 3, sc1 = (c1 & 7) * 8;
  v8h rk0, rk1, rv0, rv1;
  auto load_kv = [&](int kb) {
    const int kbase = kb * 64;
    rk0 = *(const v8h*)(Kbh + (long)(kbase + sr0) * HDIM + sc0);
    rk1 = *(const v8h*)(Kbh + (long)(kbase + sr1) * HDIM + sc1);
    rv0 = *(const v8h*)(Vbh + (long)sr0 * SEQ + kbase + sc0);
    rv1 = *(const v8h*)(Vbh + (long)sr1 * SEQ + kbase + sc1);
  };
  auto store_kv = [&](int buf) {
    *(v8h*)&Ks[buf][sr0][sc0] = rk0;
    *(v8h*)&Ks[buf][sr1][sc1] = rk1;
    *(v8h*)&Vs[buf][sr0][sc0] = rv0;
    *(v8h*)&Vs[buf][sr1][sc1] = rv1;
  };
  load_kv(0);
  store_kv(0);
#endif

  v8f o[4];
#pragma unroll
  for (int t = 0; t < 4; ++t) o[t] = {};
  v8f mi, li = {};
#pragma unroll
  for (int i = 0; i < 8; ++i) mi[i] = -3.0e38f;

  for (int kb = 0; kb < NB; ++kb) {
    const int cur = kb & 1;
#if USE_TDM
    __syncthreads();  // everyone done reading buf[1-cur] -> safe to DMA into it
    if (w == 0) {
      if (kb + 1 < NB) {
        tdm_pair((kb + 1) & 1, kb + 1);
        __builtin_amdgcn_s_wait_tensorcnt(2);  // cur pair done; next pair in flight
      } else {
        __builtin_amdgcn_s_wait_tensorcnt(0);  // last block: drain everything
      }
    }
    __syncthreads();  // cur buffer's transfers complete for all waves
#else
    if (kb + 1 < NB) load_kv(kb + 1);
    __syncthreads();
#endif

    // ---- scores: S[16 x 64] = Q(16x64) . K(64x64)^T, 4 tiles ----
    v8f s[4];
#pragma unroll
    for (int t = 0; t < 4; ++t) {
      v16h kf0 = ldfrag(&Ks[cur][t * 16 + r][kg * 8],      &Ks[cur][t * 16 + r][16 + kg * 8]);
      v16h kf1 = ldfrag(&Ks[cur][t * 16 + r][32 + kg * 8], &Ks[cur][t * 16 + r][48 + kg * 8]);
      v8f sv = {};
      sv = wmma_f16(qf0, kf0, sv);
      sv = wmma_f16(qf1, kf1, sv);
      s[t] = sv;
    }

    // ---- row max (butterfly over 16-lane halves) + online rescale ----
    v8f mnew, corr;
#pragma unroll
    for (int i = 0; i < 8; ++i) {
      float mx = fmaxf(fmaxf(s[0][i], s[1][i]), fmaxf(s[2][i], s[3][i]));
      mx = xor_max16(mx);
      mnew[i] = fmaxf(mi[i], mx);
      corr[i] = __expf(mi[i] - mnew[i]);
      mi[i] = mnew[i];
    }
    v8f p[4];
#pragma unroll
    for (int t = 0; t < 4; ++t)
#pragma unroll
      for (int i = 0; i < 8; ++i) p[t][i] = __expf(s[t][i] - mnew[i]);

    // ---- relayout P (C-fragment) -> A-fragment via wave-private LDS ----
#pragma unroll
    for (int t = 0; t < 4; ++t)
#pragma unroll
      for (int i = 0; i < 8; ++i)
        Pl[w][i + 8 * hh][t * 16 + r] = (_Float16)p[t][i];
    asm volatile("s_wait_dscnt 0" ::: "memory");  // wave-private: DS pipe is in-order
    const v16h pf0 = ldfrag(&Pl[w][r][kg * 8],      &Pl[w][r][16 + kg * 8]);
    const v16h pf1 = ldfrag(&Pl[w][r][32 + kg * 8], &Pl[w][r][48 + kg * 8]);

    // ---- row sum on the matrix pipe: rs = P . Ones ----
    v8f rs = {};
    rs = wmma_f16(pf0, ones, rs);
    rs = wmma_f16(pf1, ones, rs);
#pragma unroll
    for (int i = 0; i < 8; ++i) li[i] = li[i] * corr[i] + rs[i];
#pragma unroll
    for (int t = 0; t < 4; ++t)
#pragma unroll
      for (int i = 0; i < 8; ++i) o[t][i] *= corr[i];

    // ---- O += P(16x64) . V(64x64) from LDS Vs (d-major, keys contig) ----
#pragma unroll
    for (int t = 0; t < 4; ++t) {
      v16h vf0 = ldfrag(&Vs[cur][t * 16 + r][kg * 8],      &Vs[cur][t * 16 + r][16 + kg * 8]);
      v16h vf1 = ldfrag(&Vs[cur][t * 16 + r][32 + kg * 8], &Vs[cur][t * 16 + r][48 + kg * 8]);
      o[t] = wmma_f16(pf0, vf0, o[t]);
      o[t] = wmma_f16(pf1, vf1, o[t]);
    }

#if !USE_TDM
    if (kb + 1 < NB) {
      __syncthreads();
      store_kv((kb + 1) & 1);
    }
#endif
  }

  // ---- finalize: O /= l, write concat [B,S,E] f16 ----
#pragma unroll
  for (int i = 0; i < 8; ++i) {
    const float inv = 1.0f / li[i];
    const int s = qbase + i + 8 * hh;
    const long rowoff = ((long)b * SEQ + s) * EMBD + head * HDIM;
    Ch[rowoff + r]      = (_Float16)(o[0][i] * inv);
    Ch[rowoff + 16 + r] = (_Float16)(o[1][i] * inv);
    Ch[rowoff + 32 + r] = (_Float16)(o[2][i] * inv);
    Ch[rowoff + 48 + r] = (_Float16)(o[3][i] * inv);
  }
}

// =====================================================================
// Host launcher
// =====================================================================
extern "C" void kernel_launch(void* const* d_in, const int* in_sizes, int n_in,
                              void* d_out, int out_size, void* d_ws, size_t ws_size,
                              hipStream_t stream) {
  const float* Hf = (const float*)d_in[0];
  const float* Wq = (const float*)d_in[1];
  const float* bq = (const float*)d_in[2];
  const float* Wk = (const float*)d_in[3];
  const float* bk = (const float*)d_in[4];
  const float* Wv = (const float*)d_in[5];
  const float* bv = (const float*)d_in[6];
  const float* Wo = (const float*)d_in[7];
  const float* bo = (const float*)d_in[8];
  float* out = (float*)d_out;

  char* ws = (char*)d_ws;
  size_t cur = 0;
  auto take = [&](size_t bytes) -> void* {
    void* p = ws + cur;
    cur += (bytes + 255) & ~(size_t)255;
    return p;
  };
  _Float16* Xh   = (_Float16*)take((size_t)ROWS * EMBD * 2);
  _Float16* Wqkv = (_Float16*)take((size_t)3 * EMBD * EMBD * 2);
  _Float16* Woh  = (_Float16*)take((size_t)EMBD * EMBD * 2);
  float*    bqkv = (float*)take((size_t)3 * EMBD * 4);
  _Float16* Qh   = (_Float16*)take((size_t)BH * SEQ * HDIM * 2);
  _Float16* Khd  = (_Float16*)take((size_t)BH * SEQ * HDIM * 2);
  _Float16* Vt   = (_Float16*)take((size_t)BH * SEQ * HDIM * 2);
  _Float16* Ch   = (_Float16*)take((size_t)ROWS * EMBD * 2);

  convert_f32_f16<<<4096, 256, 0, stream>>>(Hf, Xh, ROWS * EMBD);
  pack_weights<<<1024, 256, 0, stream>>>(Wq, Wk, Wv, Wo, bq, bk, bv, Wqkv, Woh, bqkv);

  // QKV projection: M=8192, N=3072, K=1024
  gemm_wmma<0><<<dim3(ROWS / 128, (3 * EMBD) / 128), 256, 0, stream>>>(
      Xh, Wqkv, bqkv, Qh, Khd, Vt, nullptr);

  // attention: 64 (b,h) pairs x 16 q-tiles of 128
  attn_flash<<<BH * (SEQ / 128), 256, 0, stream>>>(Qh, Khd, Vt, Ch);

  // output projection: M=8192, N=1024, K=1024 -> fp32 out (+bo)
  gemm_wmma<1><<<dim3(ROWS / 128, EMBD / 128), 256, 0, stream>>>(
      Ch, Woh, bo, nullptr, nullptr, nullptr, out);
}
